// TopicAggregation_41463614275812
// MI455X (gfx1250) — matrix operand order repaired and checked by reading, not compiled, not run on hardware
//
#include <hip/hip_runtime.h>
#include <hip/hip_bf16.h>
#include <math.h>

// ---------------------------------------------------------------------------
// Shapes (compile-time constants from the reference)
// ---------------------------------------------------------------------------
#define BS     32
#define NUM_R  10
#define SL_Q   20
#define SL_H   40
#define WE     300
#define LSTM   1024
#define BILSTM 2048

typedef __attribute__((ext_vector_type(16))) __bf16          v16bf;
typedef __attribute__((ext_vector_type(8)))  float           v8f;
typedef __attribute__((ext_vector_type(8)))  unsigned short  v8u16;

// ---------------------------------------------------------------------------
// Helpers
// ---------------------------------------------------------------------------
__device__ __forceinline__ unsigned short f32_to_bf16(float f) {
    unsigned int u = __builtin_bit_cast(unsigned int, f);
    u += 0x7FFFu + ((u >> 16) & 1u);        // round-to-nearest-even
    return (unsigned short)(u >> 16);
}

// A-matrix 16x32 bf16 fragment (ISA 7.12.2 16-bit A layout):
// lane half=lane/16, m=lane%16; elems 0..7 -> K = k+half*8+e (contig),
// elems 8..15 -> K = k+16+half*8+(e-8) (contig).
__device__ __forceinline__ v16bf load_a_frag(const unsigned short* __restrict__ base,
                                             int row, int ldk, int k, int kh) {
    const v8u16 lo = *(const v8u16*)(base + (size_t)row * ldk + k + kh * 8);
    const v8u16 hi = *(const v8u16*)(base + (size_t)row * ldk + k + 16 + kh * 8);
    v16bf a;
#pragma unroll
    for (int i = 0; i < 8; ++i) {
        a[i]     = __builtin_bit_cast(__bf16, (unsigned short)lo[i]);
        a[i + 8] = __builtin_bit_cast(__bf16, (unsigned short)hi[i]);
    }
    return a;
}

// B-matrix 32x16 bf16 fragment: lane holds column n=lane%16, same K order as A
// (elems 0..7 -> K = k+half*8+e, elems 8..15 -> K = k+16+half*8+(e-8));
// source stored K-contiguous per column (i.e. W^T, N x K row-major).
__device__ __forceinline__ v16bf load_b_frag(const unsigned short* __restrict__ baseT,
                                             int col, int ldk, int k, int kh) {
    const unsigned short* p = baseT + (size_t)col * ldk + k + kh * 8;
    const v8u16 lo = *(const v8u16*)(p);
    const v8u16 hi = *(const v8u16*)(p + 16);
    v16bf b;
#pragma unroll
    for (int i = 0; i < 8; ++i) {
        b[i]     = __builtin_bit_cast(__bf16, (unsigned short)lo[i]);
        b[i + 8] = __builtin_bit_cast(__bf16, (unsigned short)hi[i]);
    }
    return b;
}

#define WMMA_BF16(a, b, c) \
    __builtin_amdgcn_wmma_f32_16x16x32_bf16(false, (a), false, (b), (short)0, (c), false, false)

// ---------------------------------------------------------------------------
// K1a: fp32 -> bf16 elementwise convert
// ---------------------------------------------------------------------------
__global__ void cvt_bf16_kernel(const float* __restrict__ in,
                                unsigned short* __restrict__ out, int n) {
    int i = blockIdx.x * blockDim.x + threadIdx.x;
    if (i < n) out[i] = f32_to_bf16(in[i]);
}

// K1b: convert (K=2048 x N=1024) fp32 weight to bf16 transposed (N x K)
__global__ void cvt_t_kernel(const float* __restrict__ in,
                             unsigned short* __restrict__ out) {
    int i = blockIdx.x * blockDim.x + threadIdx.x;   // over 2048*1024
    int k = i >> 10;
    int n = i & 1023;
    out[(size_t)n * BILSTM + k] = f32_to_bf16(in[i]);
}

// ---------------------------------------------------------------------------
// K2: gated transform GEMM:  out_bf16[M,1024] =
//     bf16( tanh(X@Wy + by) * leaky_relu(X@Wg + bg) )
// X: M x 2048 bf16, WyT/WgT: 1024 x 2048 bf16 (pre-transposed)
// Grid: (M/64, 1024/128), 256 threads = 8 waves (2M x 4N),
// wave tile 32x32 for both weight matrices: 8 WMMAs / 12 b128 loads per
// K-chunk (1.5 loads per WMMA keeps the XDL pipe fed).
// ---------------------------------------------------------------------------
__global__ __launch_bounds__(256) void gated_gemm_kernel(
    const unsigned short* __restrict__ X,
    const unsigned short* __restrict__ WyT,
    const unsigned short* __restrict__ WgT,
    const float* __restrict__ by, const float* __restrict__ bg,
    unsigned short* __restrict__ out) {
    const int lane = threadIdx.x & 31;
    const int wid  = threadIdx.x >> 5;
    const int mw   = wid & 1;                // 2 M-waves
    const int nw   = wid >> 1;               // 4 N-waves
    const int kh   = lane >> 4;
    const int l16  = lane & 15;

    const int m0 = blockIdx.x * 64 + mw * 32;
    const int n0 = blockIdx.y * 128 + nw * 32;

    const int arow0 = m0 + l16;
    const int arow1 = arow0 + 16;
    const int bcol0 = n0 + l16;
    const int bcol1 = bcol0 + 16;

    v8f aY00 = {}, aY01 = {}, aY10 = {}, aY11 = {};
    v8f aG00 = {}, aG01 = {}, aG10 = {}, aG11 = {};

    for (int k = 0; k < BILSTM; k += 32) {
        v16bf a0  = load_a_frag(X,   arow0, BILSTM, k, kh);
        v16bf a1  = load_a_frag(X,   arow1, BILSTM, k, kh);
        v16bf by0 = load_b_frag(WyT, bcol0, BILSTM, k, kh);
        v16bf by1 = load_b_frag(WyT, bcol1, BILSTM, k, kh);
        v16bf bg0 = load_b_frag(WgT, bcol0, BILSTM, k, kh);
        v16bf bg1 = load_b_frag(WgT, bcol1, BILSTM, k, kh);
        aY00 = WMMA_BF16(a0, by0, aY00);
        aY01 = WMMA_BF16(a0, by1, aY01);
        aY10 = WMMA_BF16(a1, by0, aY10);
        aY11 = WMMA_BF16(a1, by1, aY11);
        aG00 = WMMA_BF16(a0, bg0, aG00);
        aG01 = WMMA_BF16(a0, bg1, aG01);
        aG10 = WMMA_BF16(a1, bg0, aG10);
        aG11 = WMMA_BF16(a1, bg1, aG11);
    }

    // D layout: lane col N = lane%16, VGPR v -> row M = v + 8*(lane/16)
    const int oc0 = n0 + l16;
    const int oc1 = oc0 + 16;
    const float bY0 = by[oc0], bY1 = by[oc1];
    const float bG0 = bg[oc0], bG1 = bg[oc1];
#pragma unroll
    for (int v = 0; v < 8; ++v) {
        const int row0 = m0 + v + 8 * kh;
        const int row1 = row0 + 16;
        float y, g;
        y = tanhf(aY00[v] + bY0);
        g = aG00[v] + bG0; g = (g > 0.f) ? g : 0.01f * g;
        out[(size_t)row0 * LSTM + oc0] = f32_to_bf16(y * g);
        y = tanhf(aY01[v] + bY1);
        g = aG01[v] + bG1; g = (g > 0.f) ? g : 0.01f * g;
        out[(size_t)row0 * LSTM + oc1] = f32_to_bf16(y * g);
        y = tanhf(aY10[v] + bY0);
        g = aG10[v] + bG0; g = (g > 0.f) ? g : 0.01f * g;
        out[(size_t)row1 * LSTM + oc0] = f32_to_bf16(y * g);
        y = tanhf(aY11[v] + bY1);
        g = aG11[v] + bG1; g = (g > 0.f) ? g : 0.01f * g;
        out[(size_t)row1 * LSTM + oc1] = f32_to_bf16(y * g);
    }
}

// ---------------------------------------------------------------------------
// K3: score = qf @ hf^T per (b,r), mask, softmax over h -> att f32
// Block per (b*10+r), 192 threads = 6 waves covering padded 32x48 tile.
// ---------------------------------------------------------------------------
__global__ __launch_bounds__(192) void score_softmax_kernel(
    const unsigned short* __restrict__ qf,   // 640  x 1024 bf16
    const unsigned short* __restrict__ hf,   // 12800 x 1024 bf16
    const float* __restrict__ notpad,        // 32*10*40
    float* __restrict__ att) {               // 32*10*20*40
    __shared__ float s[32][48];

    const int br   = blockIdx.x;             // b*NUM_R + r
    const int b    = br / NUM_R;
    const int lane = threadIdx.x & 31;
    const int wid  = threadIdx.x >> 5;
    const int mt   = wid & 1;                // q-tile 0..1
    const int nt   = wid >> 1;               // h-tile 0..2
    const int kh   = lane >> 4;
    const int l16  = lane & 15;

    int q = mt * 16 + l16; if (q > SL_Q - 1) q = SL_Q - 1;   // clamp pad rows
    int h = nt * 16 + l16; if (h > SL_H - 1) h = SL_H - 1;   // clamp pad cols
    const int arow = b * SL_Q + q;
    const int brow = br * SL_H + h;

    v8f acc = {};
    for (int k = 0; k < LSTM; k += 32) {
        v16bf a  = load_a_frag(qf, arow, LSTM, k, kh);
        v16bf bb = load_b_frag(hf, brow, LSTM, k, kh);
        acc = WMMA_BF16(a, bb, acc);
    }

    const int hcol = nt * 16 + l16;          // 0..47
    const float mask = (hcol < SL_H) ? notpad[br * SL_H + hcol] : 0.f;
#pragma unroll
    for (int v = 0; v < 8; ++v) {
        const int qr = mt * 16 + v + 8 * kh; // 0..31
        s[qr][hcol] = acc[v] * mask + (mask - 1.f) * 10000.f;
    }
    __syncthreads();

    const int t = threadIdx.x;
    if (t < SL_Q) {
        float mx = -1e30f;
        for (int hh = 0; hh < SL_H; ++hh) mx = fmaxf(mx, s[t][hh]);
        float sum = 0.f;
        for (int hh = 0; hh < SL_H; ++hh) sum += __expf(s[t][hh] - mx);
        const float inv = 1.f / sum;
        for (int hh = 0; hh < SL_H; ++hh)
            att[(size_t)(br * SL_Q + t) * SL_H + hh] = __expf(s[t][hh] - mx) * inv;
    }
}

// ---------------------------------------------------------------------------
// K4: hist[b,q,e] = sum_r cms[b,r] * sum_h att[b,r,q,h] * embed[b,r,h,e]
// Block per (b,q), 320 threads (e < 300 active).
// ---------------------------------------------------------------------------
__global__ __launch_bounds__(320) void topic_kernel(
    const float* __restrict__ att,     // 32*10*20*40
    const float* __restrict__ embed,   // 32*10*40*300
    const float* __restrict__ cms,     // 32*10
    float* __restrict__ hist) {        // 32*20*300
    __shared__ float attl[NUM_R * SL_H];
    __shared__ float cmsl[NUM_R];

    const int bq = blockIdx.x;
    const int b = bq / SL_Q, q = bq % SL_Q;
    const int t = threadIdx.x;

    for (int i = t; i < NUM_R * SL_H; i += blockDim.x) {
        const int r = i / SL_H, hh = i % SL_H;
        attl[i] = att[(size_t)((b * NUM_R + r) * SL_Q + q) * SL_H + hh];
    }
    if (t < NUM_R) cmsl[t] = cms[b * NUM_R + t];
    __syncthreads();

    if (t < WE) {
        float acc = 0.f;
        for (int r = 0; r < NUM_R; ++r) {
            const float w = cmsl[r];
            const float* eb = embed + (size_t)((b * NUM_R + r) * SL_H) * WE + t;
            const float* al = attl + r * SL_H;
            float a2 = 0.f;
            for (int hh = 0; hh < SL_H; ++hh) a2 += al[hh] * eb[(size_t)hh * WE];
            acc += w * a2;
        }
        hist[(size_t)(b * SL_Q + q) * WE + t] = acc;
    }
}

// ---------------------------------------------------------------------------
// K5: feat = [q_embed | hist] (row of 600); out = sigmoid(feat@Wg + bg) * feat
// Block per row (640), 608 threads (n < 600 active), Wg reads coalesced in n.
// ---------------------------------------------------------------------------
__global__ __launch_bounds__(608) void gate_kernel(
    const float* __restrict__ qembed,  // 32*20*300
    const float* __restrict__ hist,    // 32*20*300
    const float* __restrict__ Wg,      // 600 x 600
    const float* __restrict__ bgv,     // 600
    float* __restrict__ out) {         // 32*20*600
    __shared__ float f[2 * WE];
    const int row = blockIdx.x;
    const int t = threadIdx.x;

    if (t < WE)           f[t] = qembed[(size_t)row * WE + t];
    else if (t < 2 * WE)  f[t] = hist[(size_t)row * WE + (t - WE)];
    __syncthreads();

    if (t < 2 * WE) {
        float acc = bgv[t];
        for (int k = 0; k < 2 * WE; ++k) acc += f[k] * Wg[(size_t)k * (2 * WE) + t];
        const float g = 1.f / (1.f + __expf(-acc));
        out[(size_t)row * (2 * WE) + t] = g * f[t];
    }
}

// ---------------------------------------------------------------------------
// Host launcher
// ---------------------------------------------------------------------------
extern "C" void kernel_launch(void* const* d_in, const int* in_sizes, int n_in,
                              void* d_out, int out_size, void* d_ws, size_t ws_size,
                              hipStream_t stream) {
    (void)in_sizes; (void)n_in; (void)out_size; (void)ws_size;

    const float* q_embed = (const float*)d_in[0];
    const float* q_enc   = (const float*)d_in[1];
    const float* h_embed = (const float*)d_in[2];
    const float* h_enc   = (const float*)d_in[3];
    const float* notpad  = (const float*)d_in[4];
    const float* cms     = (const float*)d_in[5];
    const float* Wq_y = (const float*)d_in[6];  const float* bq_y = (const float*)d_in[7];
    const float* Wq_g = (const float*)d_in[8];  const float* bq_g = (const float*)d_in[9];
    const float* Wh_y = (const float*)d_in[10]; const float* bh_y = (const float*)d_in[11];
    const float* Wh_g = (const float*)d_in[12]; const float* bh_g = (const float*)d_in[13];
    const float* Wgm  = (const float*)d_in[14]; const float* bgv  = (const float*)d_in[15];
    float* out = (float*)d_out;

    // workspace carve-up (256B aligned)
    char* ws = (char*)d_ws;
    size_t off = 0;
    auto alloc = [&](size_t bytes) -> void* {
        void* p = ws + off;
        off += (bytes + 255) & ~(size_t)255;
        return p;
    };
    const int MQ = BS * SL_Q;                   // 640
    const int MH = BS * NUM_R * SL_H;           // 12800
    unsigned short* xq  = (unsigned short*)alloc((size_t)MQ * BILSTM * 2);
    unsigned short* xh  = (unsigned short*)alloc((size_t)MH * BILSTM * 2);
    unsigned short* wqy = (unsigned short*)alloc((size_t)LSTM * BILSTM * 2);
    unsigned short* wqg = (unsigned short*)alloc((size_t)LSTM * BILSTM * 2);
    unsigned short* why = (unsigned short*)alloc((size_t)LSTM * BILSTM * 2);
    unsigned short* whg = (unsigned short*)alloc((size_t)LSTM * BILSTM * 2);
    unsigned short* qf  = (unsigned short*)alloc((size_t)MQ * LSTM * 2);
    unsigned short* hfb = (unsigned short*)alloc((size_t)MH * LSTM * 2);
    float* att  = (float*)alloc((size_t)BS * NUM_R * SL_Q * SL_H * 4);
    float* hist = (float*)alloc((size_t)BS * SL_Q * WE * 4);

    // 1) precision conversion (activations + transposed weights)
    {
        int n = MQ * BILSTM;
        cvt_bf16_kernel<<<(n + 255) / 256, 256, 0, stream>>>(q_enc, xq, n);
        n = MH * BILSTM;
        cvt_bf16_kernel<<<(n + 255) / 256, 256, 0, stream>>>(h_enc, xh, n);
        const int nt = (BILSTM * LSTM) / 256;
        cvt_t_kernel<<<nt, 256, 0, stream>>>(Wq_y, wqy);
        cvt_t_kernel<<<nt, 256, 0, stream>>>(Wq_g, wqg);
        cvt_t_kernel<<<nt, 256, 0, stream>>>(Wh_y, why);
        cvt_t_kernel<<<nt, 256, 0, stream>>>(Wh_g, whg);
    }

    // 2) gated transforms (the ~110 GFLOP bulk) on WMMA bf16
    {
        dim3 gq(MQ / 64, LSTM / 128);
        gated_gemm_kernel<<<gq, 256, 0, stream>>>(xq, wqy, wqg, bq_y, bq_g, qf);
        dim3 gh(MH / 64, LSTM / 128);
        gated_gemm_kernel<<<gh, 256, 0, stream>>>(xh, why, whg, bh_y, bh_g, hfb);
    }

    // 3) attention scores + softmax (WMMA bf16, K=1024)
    score_softmax_kernel<<<BS * NUM_R, 192, 0, stream>>>(qf, hfb, notpad, att);

    // 4) topic aggregation + context-score reduction
    topic_kernel<<<BS * SL_Q, 320, 0, stream>>>(att, h_embed, cms, hist);

    // 5) final gate
    gate_kernel<<<BS * SL_Q, 608, 0, stream>>>(q_embed, hist, Wgm, bgv, out);
}
